// PretrainedGCNKNN_85392539779214
// MI455X (gfx1250) — matrix operand order repaired and checked by reading, not compiled, not run on hardware
//
#include <hip/hip_runtime.h>
#include <hip/hip_bf16.h>
#include <stdint.h>

typedef __attribute__((ext_vector_type(16))) __bf16 v16bf;
typedef __attribute__((ext_vector_type(8)))  float  v8f;

#define N_PTS 8192
#define DIM   2048
#define HID   2048
#define OUTD  128
#define KNN_K 10
#define TOPK  11   // k+1 (rank 0 = self, dropped)

__device__ __forceinline__ unsigned short f2bf(float x) {
  unsigned int u = __float_as_uint(x);
  unsigned int r = u + 0x7FFFu + ((u >> 16) & 1u);   // round-to-nearest-even
  return (unsigned short)(r >> 16);
}

// ---- CDNA5 async global->LDS copy (ASYNCcnt path), 16B per lane ----
__device__ __forceinline__ void async_copy_b128(unsigned lds_addr, const void* gaddr) {
  asm volatile("global_load_async_to_lds_b128 %0, %1, off"
               :: "v"(lds_addr), "v"((unsigned long long)(uintptr_t)gaddr)
               : "memory");
}
__device__ __forceinline__ void wait_async() {
  asm volatile("s_wait_asynccnt 0" ::: "memory");
}

// ---------------- row l2-normalize features -> f32 + bf16 ----------------
__global__ void rownorm_kernel(const float* __restrict__ in,
                               float* __restrict__ outf,
                               unsigned short* __restrict__ outb) {
  __shared__ float red[256];
  int r = blockIdx.x, tid = threadIdx.x;
  const float* row = in + (size_t)r * DIM;
  float s = 0.f;
  for (int j = 0; j < DIM / 256; ++j) {
    float v = row[tid + j * 256];
    s += v * v;
  }
  red[tid] = s;
  __syncthreads();
  for (int st = 128; st > 0; st >>= 1) {
    if (tid < st) red[tid] += red[tid + st];
    __syncthreads();
  }
  float inv = 1.0f / fmaxf(sqrtf(red[0]), 1e-12f);
  for (int j = 0; j < DIM / 256; ++j) {
    int c = tid + j * 256;
    float v = row[c] * inv;
    outf[(size_t)r * DIM + c] = v;
    outb[(size_t)r * DIM + c] = f2bf(v);
  }
}

// ---------------- f32 -> bf16 convert (weights) ----------------
__global__ void f32_to_bf16_kernel(const float* __restrict__ in,
                                   unsigned short* __restrict__ out, int n) {
  int i = blockIdx.x * 256 + threadIdx.x;
  if (i < n) out[i] = f2bf(in[i]);
}

// ---------------- bf16 WMMA GEMM: C(MxN) = A(MxK) * B ----------------
// TRANSB=0: B is KxN row-major.  TRANSB=1: B is NxK row-major (C = A * B^T).
// M%128==0, N%128==0, K%32==0. Block = 256 threads (8 waves), 128x128 macro
// tile, each wave owns 32x64 = 2x4 v_wmma_f32_16x16x32_bf16 tiles.
// A tile (and B tile when TRANSB) are staged with global_load_async_to_lds_b128
// into double-buffered LDS; NN-B needs a K-pair pack so it stages synchronously.
template <bool TRANSB>
__global__ void gemm_bf16_kernel(const unsigned short* __restrict__ A,
                                 const unsigned short* __restrict__ B,
                                 float* __restrict__ C, int M, int N, int K) {
  __shared__ unsigned int sA[2][128][16];   // [buf][row][k-pair]
  __shared__ unsigned int sB[2][128][16];   // [buf][col][k-pair]
  const int tid  = threadIdx.x;
  const int lane = tid & 31;
  const int wid  = tid >> 5;
  const int wm   = wid >> 1;     // 0..3  (32-row slices)
  const int wn   = wid & 1;      // 0..1  (64-col slices)
  const int m0   = blockIdx.y * 128;
  const int n0   = blockIdx.x * 128;
  const int half = lane >> 4;
  const int l16  = lane & 15;

  v8f acc[2][4] = {};

  auto load_tiles = [&](int k0, int buf) {
    // A tile: 128 rows x 32 bf16 = 512 x 16B, async linear copy
#pragma unroll
    for (int i = 0; i < 2; ++i) {
      int idx = tid + i * 256;
      int row = idx >> 2, seg = idx & 3;
      async_copy_b128((unsigned)(uintptr_t)&sA[buf][row][seg * 4],
                      A + (size_t)(m0 + row) * K + k0 + seg * 8);
    }
    if (TRANSB) {
#pragma unroll
      for (int i = 0; i < 2; ++i) {
        int idx = tid + i * 256;
        int col = idx >> 2, seg = idx & 3;
        async_copy_b128((unsigned)(uintptr_t)&sB[buf][col][seg * 4],
                        B + (size_t)(n0 + col) * K + k0 + seg * 8);
      }
    } else {
      // pack B[k][n],B[k+1][n] into one dword per (col,kpair)
#pragma unroll
      for (int i = 0; i < 8; ++i) {
        int idx = tid + i * 256;               // 0..2047
        int col = idx & 127, cp = idx >> 7;    // consecutive lanes -> consecutive cols
        unsigned int lo = B[(size_t)(k0 + cp * 2)     * N + n0 + col];
        unsigned int hi = B[(size_t)(k0 + cp * 2 + 1) * N + n0 + col];
        sB[buf][col][cp] = lo | (hi << 16);
      }
    }
  };

  load_tiles(0, 0);
  wait_async();
  __syncthreads();

  int cur = 0;
  for (int k0 = 0; k0 < K; k0 += 32) {
    int nxt = cur ^ 1;
    if (k0 + 32 < K) load_tiles(k0 + 32, nxt);   // prefetch next K-tile

    // Fragments per ISA 7.12.2 layouts (16-bit A 16x32, B 32x16)
    union { unsigned int u[8]; v16bf v; } fa[2], fb[4];
#pragma unroll
    for (int r = 0; r < 8; ++r) {
      int pa = (r < 4) ? (half * 4 + r) : (8 + half * 4 + (r - 4));
      fa[0].u[r] = sA[cur][wm * 32 +      l16][pa];
      fa[1].u[r] = sA[cur][wm * 32 + 16 + l16][pa];
      int pb = pa;
#pragma unroll
      for (int tj = 0; tj < 4; ++tj)
        fb[tj].u[r] = sB[cur][wn * 64 + tj * 16 + l16][pb];
    }
#pragma unroll
    for (int ti = 0; ti < 2; ++ti)
#pragma unroll
      for (int tj = 0; tj < 4; ++tj)
        acc[ti][tj] = __builtin_amdgcn_wmma_f32_16x16x32_bf16(
            false, fa[ti].v, false, fb[tj].v, (short)0, acc[ti][tj], false, false);

    wait_async();
    __syncthreads();
    cur = nxt;
  }

#pragma unroll
  for (int ti = 0; ti < 2; ++ti)
    for (int tj = 0; tj < 4; ++tj) {
      int row0 = m0 + wm * 32 + ti * 16 + half * 8;
      int col  = n0 + wn * 64 + tj * 16 + l16;
      float* cp = C + (size_t)row0 * N + col;
#pragma unroll
      for (int r = 0; r < 8; ++r) cp[(size_t)r * N] = acc[ti][tj][r];
    }
}

// ---------------- running top-11 per row over one 2048-col Gram tile ----------------
__global__ void topk_update_kernel(const float* __restrict__ G,
                                   float* __restrict__ topv, int* __restrict__ topi,
                                   int colBase, int reset) {
  __shared__ float vals[2048];
  __shared__ float rmx[256];
  __shared__ int   rix[256];
  __shared__ float cV[TOPK];
  __shared__ int   cI[TOPK];
  int r = blockIdx.x, tid = threadIdx.x;
  for (int j = 0; j < 8; ++j)
    vals[tid + j * 256] = G[(size_t)r * 2048 + tid + j * 256];
  __syncthreads();
  for (int it = 0; it < TOPK; ++it) {
    float bv = -3.4e38f; int bi = 0;
    for (int j = 0; j < 8; ++j) {
      int c = tid + j * 256;
      float v = vals[c];
      if (v > bv) { bv = v; bi = c; }
    }
    rmx[tid] = bv; rix[tid] = bi;
    __syncthreads();
    for (int st = 128; st > 0; st >>= 1) {
      if (tid < st && rmx[tid + st] > rmx[tid]) { rmx[tid] = rmx[tid + st]; rix[tid] = rix[tid + st]; }
      __syncthreads();
    }
    if (tid == 0) {
      cV[it] = rmx[0]; cI[it] = colBase + rix[0];
      vals[rix[0]] = -3.4e38f;
    }
    __syncthreads();
  }
  if (tid == 0) {
    float* tv = topv + (size_t)r * TOPK;
    int*   tx = topi + (size_t)r * TOPK;
    if (reset) {
      for (int j = 0; j < TOPK; ++j) { tv[j] = cV[j]; tx[j] = cI[j]; }
    } else {
      float ov[TOPK]; int oi[TOPK];
      for (int j = 0; j < TOPK; ++j) { ov[j] = tv[j]; oi[j] = tx[j]; }
      int pa = 0, pb = 0;
      for (int j = 0; j < TOPK; ++j) {   // merge two descending lists, keep top 11
        if (pb >= TOPK || (pa < TOPK && ov[pa] >= cV[pb])) { tv[j] = ov[pa]; tx[j] = oi[pa]; ++pa; }
        else                                               { tv[j] = cV[pb]; tx[j] = cI[pb]; ++pb; }
      }
    }
  }
}

__global__ void knn_extract_kernel(const int* __restrict__ topi, int* __restrict__ knn) {
  int g = blockIdx.x * 256 + threadIdx.x;
  if (g < N_PTS * KNN_K) {
    int r = g / KNN_K, j = g % KNN_K;
    knn[g] = topi[r * TOPK + j + 1];   // drop rank 0 (self)
  }
}

__global__ void deg_init_kernel(float* deg) {
  int i = blockIdx.x * 256 + threadIdx.x;
  if (i < N_PTS) deg[i] = 1.0f;        // self loop
}
__global__ void deg_scatter_kernel(const int* __restrict__ knn, float* deg) {
  int g = blockIdx.x * 256 + threadIdx.x;
  if (g < N_PTS * KNN_K)
    __hip_atomic_fetch_add(&deg[knn[g]], 1.0f, __ATOMIC_RELAXED, __HIP_MEMORY_SCOPE_AGENT);
}
__global__ void dinv_kernel(const float* __restrict__ deg, float* dinv) {
  int i = blockIdx.x * 256 + threadIdx.x;
  if (i < N_PTS) dinv[i] = rsqrtf(deg[i]);
}

// out[i] = dinv[i]^2 * h[i]  (self-loop term)
__global__ void agg_init_kernel(const float* __restrict__ h, const float* __restrict__ dinv,
                                float* __restrict__ out, int C) {
  int g = blockIdx.x * 256 + threadIdx.x;
  if (g < N_PTS * C) {
    int i = g / C;
    float w = dinv[i];
    out[g] = w * w * h[g];
  }
}
// out[dst] += dinv[src]*dinv[dst]*h[src] for every (src -> dst=knn) edge
__global__ void agg_scatter_kernel(const float* __restrict__ h, const int* __restrict__ knn,
                                   const float* __restrict__ dinv, float* __restrict__ out, int C) {
  int g = blockIdx.x * 256 + threadIdx.x;
  if (g < N_PTS * KNN_K * C) {
    int e = g / C, c = g % C;
    int s = e / KNN_K;
    int d = knn[e];
    float w = dinv[s] * dinv[d];
    __hip_atomic_fetch_add(&out[(size_t)d * C + c], w * h[(size_t)s * C + c],
                           __ATOMIC_RELAXED, __HIP_MEMORY_SCOPE_AGENT);
  }
}

// z = [relu]((agg + b) * g/sqrt(1+eps) + beta) -> bf16
__global__ void bn_post_kernel(const float* __restrict__ agg, const float* __restrict__ b,
                               const float* __restrict__ g, const float* __restrict__ be,
                               unsigned short* __restrict__ zb, int C, int relu) {
  int gid = blockIdx.x * 256 + threadIdx.x;
  if (gid < N_PTS * C) {
    int c = gid % C;
    float sc = g[c] * rsqrtf(1.0f + 1e-5f);
    float v = (agg[gid] + b[c]) * sc + be[c];
    if (relu) v = fmaxf(v, 0.0f);
    zb[gid] = f2bf(v);
  }
}

// x_next = bf16(T + Sb + f)
__global__ void skip_add_kernel(const float* __restrict__ T, const float* __restrict__ Sb,
                                const float* __restrict__ f, unsigned short* __restrict__ xb) {
  int g = blockIdx.x * 256 + threadIdx.x;
  if (g < N_PTS * DIM) {
    int c = g % DIM;
    xb[g] = f2bf(T[g] + Sb[c] + f[g]);
  }
}

// out = l2norm_row(T + Sb + f)
__global__ void final_norm_kernel(const float* __restrict__ T, const float* __restrict__ Sb,
                                  const float* __restrict__ f, float* __restrict__ out) {
  __shared__ float t[2048];
  __shared__ float red[256];
  int r = blockIdx.x, tid = threadIdx.x;
  float s = 0.f;
  for (int j = 0; j < 8; ++j) {
    int c = tid + j * 256;
    float v = T[(size_t)r * DIM + c] + Sb[c] + f[(size_t)r * DIM + c];
    t[c] = v; s += v * v;
  }
  red[tid] = s;
  __syncthreads();
  for (int st = 128; st > 0; st >>= 1) {
    if (tid < st) red[tid] += red[tid + st];
    __syncthreads();
  }
  float inv = 1.0f / fmaxf(sqrtf(red[0]), 1e-12f);
  for (int j = 0; j < 8; ++j) {
    int c = tid + j * 256;
    out[(size_t)r * DIM + c] = t[c] * inv;
  }
}

extern "C" void kernel_launch(void* const* d_in, const int* in_sizes, int n_in,
                              void* d_out, int out_size, void* d_ws, size_t ws_size,
                              hipStream_t stream) {
  (void)in_sizes; (void)n_in; (void)out_size; (void)ws_size;
  const float* features = (const float*)d_in[0];
  const float* W1  = (const float*)d_in[1];
  const float* b1  = (const float*)d_in[2];
  const float* W2  = (const float*)d_in[3];
  const float* b2  = (const float*)d_in[4];
  const float* W3  = (const float*)d_in[5];
  const float* b3  = (const float*)d_in[6];
  const float* g1  = (const float*)d_in[7];
  const float* be1 = (const float*)d_in[8];
  const float* g2  = (const float*)d_in[9];
  const float* be2 = (const float*)d_in[10];
  const float* g3  = (const float*)d_in[11];
  const float* be3 = (const float*)d_in[12];
  const float* S1w = (const float*)d_in[13];
  const float* S1b = (const float*)d_in[14];
  const float* S2w = (const float*)d_in[15];
  const float* S2b = (const float*)d_in[16];
  const float* S3w = (const float*)d_in[17];
  const float* S3b = (const float*)d_in[18];

  char* ws = (char*)d_ws;
  size_t off = 0;
  auto alloc = [&](size_t bytes) -> void* {
    void* p = ws + off;
    off += (bytes + 255) & ~(size_t)255;
    return p;
  };
  float*          f    = (float*)alloc((size_t)N_PTS * DIM * 4);
  unsigned short* fb   = (unsigned short*)alloc((size_t)N_PTS * DIM * 2);
  unsigned short* xb   = (unsigned short*)alloc((size_t)N_PTS * DIM * 2);
  float*          bufA = (float*)alloc((size_t)N_PTS * DIM * 4);   // Gram tile / h / T
  float*          bufB = (float*)alloc((size_t)N_PTS * DIM * 4);   // aggregation
  unsigned short* W1b_ = (unsigned short*)alloc((size_t)DIM * HID * 2);
  unsigned short* W2b_ = (unsigned short*)alloc((size_t)HID * HID * 2);
  unsigned short* W3b_ = (unsigned short*)alloc((size_t)HID * OUTD * 2);
  unsigned short* S1wb = (unsigned short*)alloc((size_t)HID * DIM * 2);
  unsigned short* S2wb = (unsigned short*)alloc((size_t)HID * DIM * 2);
  unsigned short* S3wb = (unsigned short*)alloc((size_t)OUTD * DIM * 2);
  float* topv = (float*)alloc((size_t)N_PTS * TOPK * 4);
  int*   topi = (int*)alloc((size_t)N_PTS * TOPK * 4);
  int*   knn  = (int*)alloc((size_t)N_PTS * KNN_K * 4);
  float* deg  = (float*)alloc((size_t)N_PTS * 4);
  float* dinv = (float*)alloc((size_t)N_PTS * 4);

  // 1) l2 normalize features (f32 + bf16 copies)
  rownorm_kernel<<<dim3(N_PTS), dim3(256), 0, stream>>>(features, f, fb);

  // 2) weights -> bf16
  auto conv = [&](const float* src, unsigned short* dst, int n) {
    f32_to_bf16_kernel<<<dim3((n + 255) / 256), dim3(256), 0, stream>>>(src, dst, n);
  };
  conv(W1, W1b_, DIM * HID);
  conv(W2, W2b_, HID * HID);
  conv(W3, W3b_, HID * OUTD);
  conv(S1w, S1wb, HID * DIM);
  conv(S2w, S2wb, HID * DIM);
  conv(S3w, S3wb, OUTD * DIM);

  // 3) KNN: Gram = f@f^T in 4 column tiles of 2048 + running top-11
  //    (rows normalized => smallest distance == largest inner product)
  for (int ct = 0; ct < 4; ++ct) {
    gemm_bf16_kernel<true><<<dim3(2048 / 128, N_PTS / 128), dim3(256), 0, stream>>>(
        fb, fb + (size_t)ct * 2048 * DIM, bufA, N_PTS, 2048, DIM);
    topk_update_kernel<<<dim3(N_PTS), dim3(256), 0, stream>>>(
        bufA, topv, topi, ct * 2048, ct == 0 ? 1 : 0);
  }
  knn_extract_kernel<<<dim3((N_PTS * KNN_K + 255) / 256), dim3(256), 0, stream>>>(topi, knn);
  deg_init_kernel<<<dim3((N_PTS + 255) / 256), dim3(256), 0, stream>>>(deg);
  deg_scatter_kernel<<<dim3((N_PTS * KNN_K + 255) / 256), dim3(256), 0, stream>>>(knn, deg);
  dinv_kernel<<<dim3((N_PTS + 255) / 256), dim3(256), 0, stream>>>(deg, dinv);

  // 4) GCN layers
  auto layer = [&](const unsigned short* xin, const unsigned short* Wb, const float* bias,
                   const float* gamma, const float* beta, int Cout, int relu,
                   const unsigned short* Swb, const float* Sbias, int last) {
    // h = x @ W  (K = DIM = 2048 always: x is N x 2048)
    gemm_bf16_kernel<false><<<dim3(Cout / 128, N_PTS / 128), dim3(256), 0, stream>>>(
        xin, Wb, bufA, N_PTS, Cout, DIM);
    int totI = N_PTS * Cout;
    agg_init_kernel<<<dim3((totI + 255) / 256), dim3(256), 0, stream>>>(bufA, dinv, bufB, Cout);
    int totS = N_PTS * KNN_K * Cout;
    agg_scatter_kernel<<<dim3((totS + 255) / 256), dim3(256), 0, stream>>>(bufA, knn, dinv, bufB, Cout);
    bn_post_kernel<<<dim3((totI + 255) / 256), dim3(256), 0, stream>>>(
        bufB, bias, gamma, beta, xb, Cout, relu);
    // T = z @ Sw  (K = Cout)
    gemm_bf16_kernel<false><<<dim3(DIM / 128, N_PTS / 128), dim3(256), 0, stream>>>(
        xb, Swb, bufA, N_PTS, DIM, Cout);
    if (!last)
      skip_add_kernel<<<dim3((N_PTS * DIM + 255) / 256), dim3(256), 0, stream>>>(bufA, Sbias, f, xb);
    else
      final_norm_kernel<<<dim3(N_PTS), dim3(256), 0, stream>>>(bufA, Sbias, f, (float*)d_out);
  };
  layer(fb, W1b_, b1, g1, be1, HID,  1, S1wb, S1b, 0);
  layer(xb, W2b_, b2, g2, be2, HID,  1, S2wb, S2b, 0);
  layer(xb, W3b_, b3, g3, be3, OUTD, 0, S3wb, S3b, 1);
}
